// Attention_42391327211693
// MI455X (gfx1250) — compile-verified, matrix operand order checked by reading, not run on hardware
//
#include <hip/hip_runtime.h>

typedef _Float16 v8h  __attribute__((ext_vector_type(8)));
typedef _Float16 v16h __attribute__((ext_vector_type(16)));
typedef float    v8f  __attribute__((ext_vector_type(8)));

#define WMMA_F16(a, b, c) \
  __builtin_amdgcn_wmma_f32_16x16x32_f16(false, (a), false, (b), (short)0, (c), false, false)

// Problem constants (from reference setup_inputs)
constexpr int BB  = 2;
constexpr int SQ  = 2048;
constexpr int SK  = 2048;
constexpr int H   = 32;
constexpr int HKV = 8;
constexpr int D   = 128;
constexpr int GQA = H / HKV;   // 4

constexpr int QTILE = 128;     // q rows per block (8 waves x 16)
constexpr int KTILE = 32;      // keys per inner iteration
constexpr int NW    = QTILE / 16;  // 8 waves

// LDS layout (in halfs), strides padded, all fragment chunks 16B-aligned
constexpr int KSTR = 136;                   // K tile row stride (d-major), 272B
constexpr int VSTR = 40;                    // V^T row stride (key-major), 80B
constexpr int PSTR = 40;                    // P scratch row stride, 80B
constexpr int K_OFF = 0;
constexpr int V_OFF = KTILE * KSTR;         // 4352
constexpr int P_OFF = V_OFF + D * VSTR;     // 9472
constexpr int LDS_HALFS = P_OFF + NW * 16 * PSTR;  // 14592 halfs = 29184 B

__global__ __launch_bounds__(256)
void Attention_42391327211693_kernel(const float* __restrict__ q,
                                     const float* __restrict__ kv,
                                     const unsigned char* __restrict__ kpmask,
                                     float* __restrict__ out) {
  __shared__ __align__(16) _Float16 smem[LDS_HALFS];

  const int tid  = threadIdx.x;
  const int lane = tid & 31;
  const int wid  = tid >> 5;        // 0..7
  const int half = lane >> 4;       // 0: lanes 0-15, 1: lanes 16-31
  const int lan  = lane & 15;

  const int qb  = blockIdx.x * QTILE;
  const int h   = blockIdx.y;
  const int bb  = blockIdx.z;
  const int hkv = h / GQA;
  const float scale = 0.08838834764831845f;  // 1/sqrt(128)

  // ---- Q A-fragments (16x32, K-dim = d), scale folded in ----
  // A layout: lane row m = lan; lanes 0-15 hold k = {0..7,16..23}, lanes 16-31 k = {8..15,24..31}
  const int qrowA = qb + wid * 16 + lan;
  const float* qp = q + (((long)bb * SQ + qrowA) * H + h) * D;
  v16h aq[4];
#pragma unroll
  for (int f = 0; f < 4; ++f) {
#pragma unroll
    for (int i = 0; i < 8; ++i) {
      aq[f][i]     = (_Float16)(qp[f * 32 + 8 * half + i] * scale);
      aq[f][8 + i] = (_Float16)(qp[f * 32 + 16 + 8 * half + i] * scale);
    }
  }

  // ---- Accumulators and softmax state ----
  v8f acc[8];
#pragma unroll
  for (int dt = 0; dt < 8; ++dt)
#pragma unroll
    for (int i = 0; i < 8; ++i) acc[dt][i] = 0.0f;

  float mrow[8], lrow[8];
#pragma unroll
  for (int r = 0; r < 8; ++r) { mrow[r] = -1e30f; lrow[r] = 0.0f; }

  const int kbmax = qb + QTILE - KTILE;  // last key block (causal frontier)

  // Staging split: 256 threads, each owns 16 d-elements of one of 32 K/V rows.
  const int sr  = tid >> 3;          // key row 0..31
  const int sd0 = (tid & 7) * 16;    // d chunk

  float kreg[16], vreg[16];

  // Global -> registers for tile kb (issued early; consumed at next store).
  auto load_tile = [&](int kb) {
    const float* ks = kv + ((((long)bb * SK + kb + sr) * 2 + 0) * HKV + hkv) * D + sd0;
    const float* vs = kv + ((((long)bb * SK + kb + sr) * 2 + 1) * HKV + hkv) * D + sd0;
#pragma unroll
    for (int j = 0; j < 16; ++j) { kreg[j] = ks[j]; vreg[j] = vs[j]; }
    if (kb + KTILE <= kbmax) {  // hint the tile after next into GL2
      __builtin_prefetch(ks + (long)KTILE * 2 * HKV * D, 0, 1);
      __builtin_prefetch(vs + (long)KTILE * 2 * HKV * D, 0, 1);
    }
  };

  // Registers -> LDS: K row-major [key][d] f16, V transposed [d][key] f16.
  auto store_tile = [&]() {
#pragma unroll
    for (int j = 0; j < 16; ++j)
      smem[K_OFF + sr * KSTR + sd0 + j] = (_Float16)kreg[j];
#pragma unroll
    for (int j = 0; j < 16; ++j)
      smem[V_OFF + (sd0 + j) * VSTR + sr] = (_Float16)vreg[j];
  };

  // One key-block compute step (LDS tile must be ready); do_causal folds away.
  auto compute_step = [&](int kb, bool do_causal) {
    // ---- GEMM1: S = Q @ K^T; preload 4 B-fragments, then 4 WMMAs ----
    float sv[2][8];
#pragma unroll
    for (int nt = 0; nt < 2; ++nt) {
      const _Float16* kbase = smem + K_OFF + (nt * 16 + lan) * KSTR;  // key = col N
      v16h bf[4];
#pragma unroll
      for (int f = 0; f < 4; ++f) {
        v8h lo = *(const v8h*)(kbase + f * 32 + 8 * half);
        v8h hi = *(const v8h*)(kbase + f * 32 + 16 + 8 * half);
        bf[f] = __builtin_shufflevector(lo, hi, 0, 1, 2, 3, 4, 5, 6, 7,
                                        8, 9, 10, 11, 12, 13, 14, 15);
      }
      v8f c;
#pragma unroll
      for (int i = 0; i < 8; ++i) c[i] = 0.0f;
#pragma unroll
      for (int f = 0; f < 4; ++f) c = WMMA_F16(aq[f], bf[f], c);

      const int key  = kb + nt * 16 + lan;
      const float pb = kpmask[bb * SK + key] ? 0.0f : -10000.0f;
#pragma unroll
      for (int r = 0; r < 8; ++r) {
        float s = c[r] + pb;
        if (do_causal) {
          const int grow = qb + wid * 16 + 8 * half + r;  // C-layout row
          if (key > grow) s = -10000.0f;                  // causal (Sk == Sq)
        }
        sv[nt][r] = s;
      }
    }

    // ---- Online softmax (row reductions across 16-lane halves) ----
    float corr[8];
#pragma unroll
    for (int r = 0; r < 8; ++r) {
      float mr = fmaxf(sv[0][r], sv[1][r]);
      mr = fmaxf(mr, __shfl_xor(mr, 1));
      mr = fmaxf(mr, __shfl_xor(mr, 2));
      mr = fmaxf(mr, __shfl_xor(mr, 4));
      mr = fmaxf(mr, __shfl_xor(mr, 8));
      const float mn = fmaxf(mrow[r], mr);
      corr[r] = __expf(mrow[r] - mn);
      mrow[r] = mn;
      const float p0 = __expf(sv[0][r] - mn);
      const float p1 = __expf(sv[1][r] - mn);
      sv[0][r] = p0;
      sv[1][r] = p1;
      float rs = p0 + p1;
      rs += __shfl_xor(rs, 1);
      rs += __shfl_xor(rs, 2);
      rs += __shfl_xor(rs, 4);
      rs += __shfl_xor(rs, 8);
      lrow[r] = lrow[r] * corr[r] + rs;
    }
#pragma unroll
    for (int dt = 0; dt < 8; ++dt)
#pragma unroll
      for (int r = 0; r < 8; ++r) acc[dt][r] *= corr[r];

    // ---- P: C layout -> A layout via per-wave LDS bounce ----
    _Float16* pl = smem + P_OFF + wid * 16 * PSTR;
#pragma unroll
    for (int nt = 0; nt < 2; ++nt)
#pragma unroll
      for (int r = 0; r < 8; ++r)
        pl[(8 * half + r) * PSTR + nt * 16 + lan] = (_Float16)sv[nt][r];
    asm volatile("s_wait_dscnt 0" ::: "memory");

    v16h pa;
    {
      const _Float16* pr = pl + lan * PSTR;
      v8h lo = *(const v8h*)(pr + 8 * half);
      v8h hi = *(const v8h*)(pr + 16 + 8 * half);
      pa = __builtin_shufflevector(lo, hi, 0, 1, 2, 3, 4, 5, 6, 7,
                                   8, 9, 10, 11, 12, 13, 14, 15);
    }

    // ---- GEMM2: O += P @ V; two groups of (4 preloads + 4 WMMAs) ----
#pragma unroll
    for (int g = 0; g < 2; ++g) {
      v16h bv[4];
#pragma unroll
      for (int j = 0; j < 4; ++j) {
        const int dt = g * 4 + j;
        const _Float16* vb = smem + V_OFF + (dt * 16 + lan) * VSTR;  // d = col N
        v8h lo = *(const v8h*)(vb + 8 * half);
        v8h hi = *(const v8h*)(vb + 16 + 8 * half);
        bv[j] = __builtin_shufflevector(lo, hi, 0, 1, 2, 3, 4, 5, 6, 7,
                                        8, 9, 10, 11, 12, 13, 14, 15);
      }
#pragma unroll
      for (int j = 0; j < 4; ++j)
        acc[g * 4 + j] = WMMA_F16(pa, bv[j], acc[g * 4 + j]);
    }
  };

  // ---- Software-pipelined main loop: global loads for tile k+1 in flight
  //      while tile k's 16 WMMAs execute. ----
  load_tile(0);
  for (int kb = 0; kb < qb; kb += KTILE) {       // keys strictly below q tile
    store_tile();
    __syncthreads();
    load_tile(kb + KTILE);                       // always valid here (kb+32 <= qb <= kbmax)
    compute_step(kb, false);
    __syncthreads();
  }
  for (int kb = qb; kb <= kbmax; kb += KTILE) {  // causal frontier tiles
    store_tile();
    __syncthreads();
    if (kb + KTILE <= kbmax) load_tile(kb + KTILE);
    compute_step(kb, true);
    __syncthreads();
  }

  // ---- Epilogue: normalize and store (out[b][t][h][d], fp32) ----
#pragma unroll
  for (int r = 0; r < 8; ++r) {
    const float inv = 1.0f / lrow[r];
    const int grow = qb + wid * 16 + 8 * half + r;
    float* op = out + (((long)bb * SQ + grow) * H + h) * D + lan;
#pragma unroll
    for (int dt = 0; dt < 8; ++dt) op[dt * 16] = acc[dt][r] * inv;
  }
}

extern "C" void kernel_launch(void* const* d_in, const int* in_sizes, int n_in,
                              void* d_out, int out_size, void* d_ws, size_t ws_size,
                              hipStream_t stream) {
  (void)in_sizes; (void)n_in; (void)out_size; (void)d_ws; (void)ws_size;
  const float* q  = (const float*)d_in[0];
  const float* kv = (const float*)d_in[1];
  const unsigned char* kpmask = (const unsigned char*)d_in[2];  // jnp bool = 1 byte
  float* out = (float*)d_out;

  dim3 grid(SQ / QTILE, H, BB);  // (16, 32, 2)
  dim3 block(256);               // 8 waves (wave32)
  Attention_42391327211693_kernel<<<grid, block, 0, stream>>>(q, kv, kpmask, out);
}
